// DAC_structure_64914135712466
// MI455X (gfx1250) — compile-verified, matrix operand order checked by reading, not run on hardware
//
#include <hip/hip_runtime.h>

typedef __attribute__((ext_vector_type(16))) _Float16 v16h;
typedef __attribute__((ext_vector_type(8)))  float    v8f;

#define B_    32
#define L_    128
#define H_    8
#define E_    64
#define P_    16            // patch size -> batch-group size
#define G_    (B_ / P_)     // 2 groups
#define SCALE 0.125f        // 1/sqrt(64)
#define QS    72            // padded LDS stride (halfs) for Q/K/V tiles
#define SS    132           // padded LDS stride for S (f32) and P (f16)

// ---------------- zero the mean-accumulator buffer (1 MB in d_ws) -------------
__global__ void zero_kernel(float4* __restrict__ p, int n4) {
    int i = blockIdx.x * blockDim.x + threadIdx.x;
    if (i < n4) p[i] = make_float4(0.f, 0.f, 0.f, 0.f);
}

// ---------------- attention per (b,h): S=softmax(QK^T/8), O=S*V ---------------
__global__ __launch_bounds__(256) void attn_kernel(
    const float* __restrict__ q, const float* __restrict__ k,
    const float* __restrict__ v, float* __restrict__ outV,
    float* __restrict__ meanbuf)
{
    __shared__ _Float16 Qh[L_][QS];     // Q tile, f16
    __shared__ _Float16 KVh[L_][QS];    // K tile, later reloaded with V
    __shared__ float    S[L_][SS];      // scores / exp row buffer
    __shared__ _Float16 Ph[L_][SS];     // softmax probs, f16

    const int bh   = blockIdx.x;
    const int b    = bh / H_;
    const int h    = bh % H_;
    const int tid  = threadIdx.x;
    const int wave = tid >> 5;
    const int lane = tid & 31;
    const int lm   = lane & 15;         // N / M position within half-wave
    const int hi   = lane >> 4;         // lane-group select
    const int row0 = wave * 16;         // this wave's 16-row stripe

    const size_t base = ((size_t)(b * L_) * H_ + h) * E_;   // row stride = H_*E_ = 512
    const float* Qg = q + base;
    const float* Kg = k + base;
    const float* Vg = v + base;

    // ---- stage Q,K into LDS as f16 (float4-vectorized global reads) ----
    for (int i = tid; i < L_ * E_ / 4; i += 256) {
        int l  = i >> 4;
        int e4 = (i & 15) * 4;
        float4 fq = *(const float4*)(Qg + (size_t)l * 512 + e4);
        float4 fk = *(const float4*)(Kg + (size_t)l * 512 + e4);
        Qh[l][e4 + 0] = (_Float16)fq.x;  Qh[l][e4 + 1] = (_Float16)fq.y;
        Qh[l][e4 + 2] = (_Float16)fq.z;  Qh[l][e4 + 3] = (_Float16)fq.w;
        KVh[l][e4 + 0] = (_Float16)fk.x; KVh[l][e4 + 1] = (_Float16)fk.y;
        KVh[l][e4 + 2] = (_Float16)fk.z; KVh[l][e4 + 3] = (_Float16)fk.w;
    }
    __syncthreads();

    // ---- phase A: S[row0..row0+15][0..127] = SCALE * Q K^T via WMMA ----
    #pragma unroll
    for (int n = 0; n < 8; ++n) {
        v8f c = {};
        #pragma unroll
        for (int kc = 0; kc < 2; ++kc) {
            v16h a, bb;
            #pragma unroll
            for (int t = 0; t < 16; ++t) {
                // ISA 16-bit A 16x32 layout: lanes0-15 K0-7/K16-23, lanes16-31 K8-15/K24-31
                int kk = (t < 8) ? (hi * 8 + t) : (16 + hi * 8 + (t - 8));
                a[t]  = Qh[row0 + lm][kc * 32 + kk];            // A = Q rows
                bb[t] = KVh[n * 16 + lm][kc * 32 + kk];         // B = K^T (col = K row)
            }
            c = __builtin_amdgcn_wmma_f32_16x16x32_f16(false, a, false, bb,
                                                       (short)0, c, false, false);
        }
        #pragma unroll
        for (int r = 0; r < 8; ++r)                              // D: vgpr r -> M = r + 8*hi
            S[row0 + r + 8 * hi][n * 16 + lm] = SCALE * c[r];
    }
    __syncthreads();

    // ---- reload V into KVh (K no longer needed) ----
    for (int i = tid; i < L_ * E_ / 4; i += 256) {
        int l  = i >> 4;
        int e4 = (i & 15) * 4;
        float4 fv = *(const float4*)(Vg + (size_t)l * 512 + e4);
        KVh[l][e4 + 0] = (_Float16)fv.x; KVh[l][e4 + 1] = (_Float16)fv.y;
        KVh[l][e4 + 2] = (_Float16)fv.z; KVh[l][e4 + 3] = (_Float16)fv.w;
    }

    // ---- phase B: row softmax (threads 0..127), accumulate group mean ----
    if (tid < L_) {
        const int row = tid;
        float m = -1e30f;
        for (int j = 0; j < L_; ++j) m = fmaxf(m, S[row][j]);
        float s = 0.f;
        for (int j = 0; j < L_; ++j) { float e = __expf(S[row][j] - m); S[row][j] = e; s += e; }
        const float inv = 1.f / s;
        float* mrow = meanbuf + (((size_t)(b / P_) * H_ + h) * L_ + row) * L_;
        for (int j = 0; j < L_; ++j) {
            float pr = S[row][j] * inv;
            Ph[row][j] = (_Float16)pr;
            atomicAdd(&mrow[j], pr * (1.0f / P_));   // mean over batch group of 16
        }
    }
    __syncthreads();

    // ---- phase C: O[16 x 64] = P[16 x 128] * V[128 x 64] via WMMA ----
    #pragma unroll
    for (int n = 0; n < 4; ++n) {
        v8f c = {};
        #pragma unroll
        for (int kc = 0; kc < 4; ++kc) {
            v16h a, bb;
            #pragma unroll
            for (int t = 0; t < 16; ++t) {
                int kk = (t < 8) ? (hi * 8 + t) : (16 + hi * 8 + (t - 8));
                a[t]  = Ph[row0 + lm][kc * 32 + kk];            // A = P rows
                bb[t] = KVh[kc * 32 + kk][n * 16 + lm];         // B = V (row-major)
            }
            c = __builtin_amdgcn_wmma_f32_16x16x32_f16(false, a, false, bb,
                                                       (short)0, c, false, false);
        }
        #pragma unroll
        for (int r = 0; r < 8; ++r) {
            int row = row0 + r + 8 * hi;
            outV[(((size_t)b * L_ + row) * H_ + h) * E_ + n * 16 + lm] = c[r];
        }
    }
}

// ---------------- tile/broadcast mean [2,8,128,128] -> [2,8,2048,2048] --------
// Pure HBM-write-bound: 268 MB of float4 stores; 1 MB source stays in L2.
__global__ __launch_bounds__(256) void tile_kernel(const float* __restrict__ mean,
                                                   float4* __restrict__ out)
{
    const long long N4 = (long long)G_ * H_ * 2048LL * 512LL;   // 16,777,216 float4
    for (long long idx = (long long)blockIdx.x * 256 + threadIdx.x;
         idx < N4; idx += (long long)gridDim.x * 256) {
        int j4 = (int)(idx & 511);          // float4 column in [0,512)
        int i  = (int)((idx >> 9) & 2047);  // row in [0,2048)
        int gh = (int)(idx >> 20);          // (g,h) in [0,16)
        const float4* src =
            (const float4*)(mean + ((size_t)gh * L_ + (i & 127)) * L_) + (j4 & 31);
        out[idx] = *src;
    }
}

extern "C" void kernel_launch(void* const* d_in, const int* in_sizes, int n_in,
                              void* d_out, int out_size, void* d_ws, size_t ws_size,
                              hipStream_t stream) {
    (void)in_sizes; (void)n_in; (void)out_size; (void)ws_size;
    const float* q = (const float*)d_in[0];
    const float* k = (const float*)d_in[1];
    const float* v = (const float*)d_in[2];
    float* outV  = (float*)d_out;                                  // [32,128,8,64]
    float* outS  = (float*)d_out + (size_t)B_ * L_ * H_ * E_;      // [2,8,2048,2048]
    float* mean  = (float*)d_ws;                                   // [2,8,128,128] = 1 MB

    const int n4 = G_ * H_ * L_ * L_ / 4;                          // 65536
    zero_kernel<<<(n4 + 255) / 256, 256, 0, stream>>>((float4*)mean, n4);
    attn_kernel<<<B_ * H_, 256, 0, stream>>>(q, k, v, outV, mean);
    tile_kernel<<<8192, 256, 0, stream>>>(mean, (float4*)outS);
}
